// GRULayer_85426899517829
// MI455X (gfx1250) — matrix operand order repaired and checked by reading, not compiled, not run on hardware
//
#include <hip/hip_runtime.h>
#include <hip/hip_bf16.h>

// GRU layer for MI455X (gfx1250), wave32 + v_wmma_f32_16x16x32_bf16.
//
// Decomposition:
//  - Batch rows are independent across the recurrence -> 16 workgroups,
//    each owning a 16-row batch chunk, h resident in LDS for all 512 steps.
//  - 16 waves per workgroup; wave w owns hidden columns [16w, 16w+16).
//  - Per step, per wave: 48 WMMAs (16x16x32 bf16, f32 accum).
//  - Weights pre-packed into a per-lane B-fragment layout (contiguous k = 16*hi + e); the pack pointer is
//    laundered through an empty asm each timestep so the fragment loads are
//    NOT hoisted out of the t-loop (which previously caused ~384 VGPRs of
//    scratch spill). Weights stay L2-resident (768 KB) and are re-streamed.

typedef __attribute__((ext_vector_type(8)))  __bf16 v8bf;
typedef __attribute__((ext_vector_type(16))) __bf16 v16bf;
typedef __attribute__((ext_vector_type(8)))  float  v8f;

#define TT 512
#define BB 256
#define IC 256
#define HH 256
#define ROW 264   // 256 + 8 bf16 pad -> row stride 528 B (16B aligned, rotates banks)

static __device__ inline v16bf join8(v8bf a, v8bf b) {
    return __builtin_shufflevector(a, b, 0,1,2,3,4,5,6,7,8,9,10,11,12,13,14,15);
}

// A-fragment (16x32 bf16) from LDS, row-major rows of ROW bf16.
// Lanes 0-15 (hi=0): row m, K = {k0..k0+15}
// Lanes 16-31 (hi=1): row m, K = {k0+16..k0+31}
static __device__ inline v16bf ldA(const __bf16* base, int m, int hi, int k0) {
    const __bf16* p = base + m * ROW + k0 + hi * 16;
    v8bf lo = *reinterpret_cast<const v8bf*>(p);        // ds_load_b128
    v8bf hh = *reinterpret_cast<const v8bf*>(p + 8);    // ds_load_b128
    return join8(lo, hh);
}

// B-fragment (32x16 bf16): packed per-lane contiguous by the prep kernel,
// so a wave reads 1024 contiguous bytes (2x global_load_b128 per lane).
static __device__ inline v16bf ldB(const __bf16* __restrict__ pack,
                                   int g, int w, int kc, int lane) {
    const __bf16* p = pack + (size_t)(((g * 16 + w) * 8) + kc) * 512 + lane * 16;
    v8bf lo = *reinterpret_cast<const v8bf*>(p);
    v8bf hh = *reinterpret_cast<const v8bf*>(p + 8);
    return join8(lo, hh);
}

static __device__ inline v8f wmma_bf16(v16bf a, v16bf b, v8f c) {
    return __builtin_amdgcn_wmma_f32_16x16x32_bf16(
        /*neg_a=*/false, a, /*neg_b=*/false, b,
        /*c_mod=*/(short)0, c, /*reuse_a=*/false, /*reuse_b=*/false);
}

// ---------------------------------------------------------------------------
// Prep: convert 6 f32 weight matrices [256,256] to bf16 in B-fragment order.
// g: 0=Wz_x 1=Wz_h 2=Wr_x 3=Wr_h 4=Wh_x 5=Wh_h
// pack[((g*16+nt)*8+kc)*512 + lane*16 + e] = W[nt*16 + (lane&15)][kc*32 + (lane>>4)*16 + e]
// ---------------------------------------------------------------------------
__global__ void gru_pack_weights(const float* __restrict__ Wz_x, const float* __restrict__ Wz_h,
                                 const float* __restrict__ Wr_x, const float* __restrict__ Wr_h,
                                 const float* __restrict__ Wh_x, const float* __restrict__ Wh_h,
                                 __bf16* __restrict__ pack) {
    int gid = blockIdx.x * blockDim.x + threadIdx.x;   // 0 .. 6*65536-1
    int g   = gid >> 16;
    int rem = gid & 0xFFFF;
    int e    = rem & 15;
    int lane = (rem >> 4) & 31;
    int kc   = (rem >> 9) & 7;
    int nt   = rem >> 12;
    int n  = lane & 15;
    int hi = lane >> 4;
    int k = kc * 32 + hi * 16 + e;
    int j = nt * 16 + n;
    const float* src;
    if      (g == 0) src = Wz_x;
    else if (g == 1) src = Wz_h;
    else if (g == 2) src = Wr_x;
    else if (g == 3) src = Wr_h;
    else if (g == 4) src = Wh_x;
    else             src = Wh_h;
    pack[gid] = (__bf16)src[j * IC + k];
}

__global__ void gru_pack_bias(const float* __restrict__ bz_x, const float* __restrict__ bz_h,
                              const float* __restrict__ br_x, const float* __restrict__ br_h,
                              const float* __restrict__ bh_x, const float* __restrict__ bh_h,
                              float* __restrict__ bias) {
    int j = threadIdx.x;
    if (j < HH) {
        bias[j]          = bz_x[j] + bz_h[j];
        bias[HH + j]     = br_x[j] + br_h[j];
        bias[2 * HH + j] = bh_x[j] + bh_h[j];
    }
}

// ---------------------------------------------------------------------------
// Persistent scan: 16 blocks x 512 threads (16 waves). Block owns batch rows
// [16*blockIdx, 16*blockIdx+16); wave w owns hidden columns [16w, 16w+16).
// ---------------------------------------------------------------------------
__global__ __launch_bounds__(512, 1)
void gru_scan(const float* __restrict__ x, const __bf16* __restrict__ pack,
              const float* __restrict__ bias, float* __restrict__ out) {
    __shared__ __align__(16) __bf16 lds_x [16 * ROW];  // x_t chunk (bf16)
    __shared__ __align__(16) __bf16 lds_h [16 * ROW];  // hidden state (bf16)
    __shared__ __align__(16) __bf16 lds_r [16 * ROW];  // reset gate
    __shared__ __align__(16) __bf16 lds_rh[16 * ROW];  // r * h

    const int tid  = threadIdx.x;
    const int lane = tid & 31;
    const int w    = tid >> 5;        // wave id 0..15 -> N tile
    const int n    = lane & 15;       // also A-fragment row m
    const int hi   = lane >> 4;
    const int b0   = blockIdx.x * 16; // batch chunk base
    const int ncol = w * 16 + n;      // this lane's output column

    const float bz = bias[ncol];
    const float br = bias[HH + ncol];
    const float bh = bias[2 * HH + ncol];

    // h0 = 0
    for (int i = tid; i < 16 * ROW; i += 512) lds_h[i] = (__bf16)0.0f;
    __syncthreads();

    // x staging: thread handles 8 contiguous elems of the 16x256 tile
    const int sm = tid >> 5;            // row 0..15
    const int sk = (tid & 31) * 8;      // col base 0..248

    float* hlast = out + (size_t)TT * BB * HH;

    for (int t = 0; t < TT; ++t) {
        // Launder the weight-pack base each iteration so LICM cannot hoist
        // the 48 B-fragments (384 VGPRs) out of the loop and spill them to
        // scratch. poff is always 0 but opaque to the optimizer; the loads
        // re-issue every step and hit L2 (weights are only 768 KB).
        unsigned poff = 0;
        asm volatile("" : "+s"(poff));
        const __bf16* __restrict__ pk = pack + poff;

        // ---- stage x_t (f32 -> bf16), prefetch next step's tile ----
        {
            const float* xp = x + ((size_t)t * BB + b0 + sm) * IC + sk;
            float4 f0 = *reinterpret_cast<const float4*>(xp);
            float4 f1 = *reinterpret_cast<const float4*>(xp + 4);
            __bf16* dp = &lds_x[sm * ROW + sk];
            dp[0] = (__bf16)f0.x; dp[1] = (__bf16)f0.y;
            dp[2] = (__bf16)f0.z; dp[3] = (__bf16)f0.w;
            dp[4] = (__bf16)f1.x; dp[5] = (__bf16)f1.y;
            dp[6] = (__bf16)f1.z; dp[7] = (__bf16)f1.w;
            if (t + 1 < TT) __builtin_prefetch(xp + BB * IC, 0, 1); // global_prefetch_b8
        }
        __syncthreads();

        // ---- phase 0: z and r gate pre-activations (32 WMMAs) ----
        v8f az = {}; v8f ar = {};
        #pragma unroll
        for (int kc = 0; kc < 8; ++kc) {
            v16bf a_x = ldA(lds_x, n, hi, kc * 32);
            v16bf a_h = ldA(lds_h, n, hi, kc * 32);
            v16bf bzx = ldB(pk, 0, w, kc, lane);
            v16bf bzh = ldB(pk, 1, w, kc, lane);
            v16bf brx = ldB(pk, 2, w, kc, lane);
            v16bf brh = ldB(pk, 3, w, kc, lane);
            az = wmma_bf16(a_x, bzx, az);
            az = wmma_bf16(a_h, bzh, az);
            ar = wmma_bf16(a_x, brx, ar);
            ar = wmma_bf16(a_h, brh, ar);
        }
        // activations; keep z in regs, publish r to LDS.
        // C/D layout: VGPR p -> M = p + 8*hi, N = lane&15.
        float zg[8];
        #pragma unroll
        for (int p = 0; p < 8; ++p) {
            zg[p]    = 1.0f / (1.0f + __expf(-(az[p] + bz)));
            float rg = 1.0f / (1.0f + __expf(-(ar[p] + br)));
            int M = p + hi * 8;
            lds_r[M * ROW + ncol] = (__bf16)rg;
        }
        __syncthreads();

        // ---- elementwise r * h ----
        #pragma unroll
        for (int q = 0; q < 8; ++q) {
            int off = sm * ROW + sk + q;
            lds_rh[off] = (__bf16)((float)lds_r[off] * (float)lds_h[off]);
        }
        __syncthreads();

        // ---- phase 1: candidate pre-activation (16 WMMAs) ----
        v8f ah = {};
        #pragma unroll
        for (int kc = 0; kc < 8; ++kc) {
            v16bf a_x  = ldA(lds_x,  n, hi, kc * 32);
            v16bf a_rh = ldA(lds_rh, n, hi, kc * 32);
            v16bf bhx  = ldB(pk, 4, w, kc, lane);
            v16bf bhh  = ldB(pk, 5, w, kc, lane);
            ah = wmma_bf16(a_x,  bhx, ah);
            ah = wmma_bf16(a_rh, bhh, ah);
        }
        // ---- gated update, write state + streamed output ----
        #pragma unroll
        for (int p = 0; p < 8; ++p) {
            int M = p + hi * 8;
            float htil = tanhf(ah[p] + bh);
            float hold = (float)lds_h[M * ROW + ncol];
            float hnew = (1.0f - zg[p]) * hold + zg[p] * htil;
            lds_h[M * ROW + ncol] = (__bf16)hnew;
            out[((size_t)t * BB + (b0 + M)) * HH + ncol] = hnew;
            if (t == TT - 1) hlast[(size_t)(b0 + M) * HH + ncol] = hnew;
        }
        __syncthreads();
    }
}

extern "C" void kernel_launch(void* const* d_in, const int* in_sizes, int n_in,
                              void* d_out, int out_size, void* d_ws, size_t ws_size,
                              hipStream_t stream) {
    const float* x    = (const float*)d_in[0];
    const float* Wz_x = (const float*)d_in[1];  const float* bz_x = (const float*)d_in[2];
    const float* Wz_h = (const float*)d_in[3];  const float* bz_h = (const float*)d_in[4];
    const float* Wr_x = (const float*)d_in[5];  const float* br_x = (const float*)d_in[6];
    const float* Wr_h = (const float*)d_in[7];  const float* br_h = (const float*)d_in[8];
    const float* Wh_x = (const float*)d_in[9];  const float* bh_x = (const float*)d_in[10];
    const float* Wh_h = (const float*)d_in[11]; const float* bh_h = (const float*)d_in[12];

    __bf16* pack = (__bf16*)d_ws;                                   // 6*65536 bf16 = 768 KB
    float*  bias = (float*)((char*)d_ws + (size_t)6 * 65536 * 2);   // 3*256 f32

    gru_pack_weights<<<768, 512, 0, stream>>>(Wz_x, Wz_h, Wr_x, Wr_h, Wh_x, Wh_h, pack);
    gru_pack_bias<<<1, 256, 0, stream>>>(bz_x, bz_h, br_x, br_h, bh_x, bh_h, bias);
    gru_scan<<<16, 512, 0, stream>>>(x, pack, bias, (float*)d_out);
}
